// EnhancedCNNIntegrator_30236569764100
// MI455X (gfx1250) — compile-verified, matrix operand order checked by reading, not run on hardware
//
#include <hip/hip_runtime.h>

#define Hdim 1024
#define NHEAD 16
#define HDh 64
#define Bb 4
#define Ss 1024
#define SAa 1027
#define SPp 1056   // padded seq (multiple of 32)

typedef __attribute__((ext_vector_type(16))) __bf16 v16bf;
typedef __attribute__((ext_vector_type(8)))  __bf16 v8bf;
typedef __attribute__((ext_vector_type(8)))  float  v8f;
typedef __attribute__((ext_vector_type(4)))  int    v4i;

#if defined(__has_builtin)
#if __has_builtin(__builtin_amdgcn_global_load_async_to_lds_b128) && \
    __has_builtin(__builtin_amdgcn_s_wait_asynccnt)
#define USE_ASYNC_LDS 1
#endif
#endif

// Load a 16-element bf16 WMMA fragment for one lane.
// Per CDNA5 16-bit A/B layout: elems 0..7 -> K = half*8 + j, elems 8..15 -> K = 16 + half*8 + j.
static __device__ inline v16bf load_frag(const __bf16* base, int hl) {
  v8bf lo = *reinterpret_cast<const v8bf*>(base + hl * 8);
  v8bf hi = *reinterpret_cast<const v8bf*>(base + 16 + hl * 8);
  v16bf r;
#pragma unroll
  for (int i = 0; i < 8; ++i) { r[i] = lo[i]; r[i + 8] = hi[i]; }
  return r;
}

static __device__ inline v8f wmma_bf(v16bf a, v16bf b, v8f c) {
  return __builtin_amdgcn_wmma_f32_16x16x32_bf16(false, a, false, b, (short)0, c, false, false);
}

// Butterfly reductions over the 16 lanes of a half-wave (rows live per-half in C layout).
static __device__ inline float rmax16(float v) {
  v = fmaxf(v, __shfl_xor(v, 1, 32));
  v = fmaxf(v, __shfl_xor(v, 2, 32));
  v = fmaxf(v, __shfl_xor(v, 4, 32));
  v = fmaxf(v, __shfl_xor(v, 8, 32));
  return v;
}
static __device__ inline float rsum16(float v) {
  v += __shfl_xor(v, 1, 32);
  v += __shfl_xor(v, 2, 32);
  v += __shfl_xor(v, 4, 32);
  v += __shfl_xor(v, 8, 32);
  return v;
}

// ---------------- pack: f32 features + physics rows -> padded bf16 [B][SP][H] ----------------
__global__ void pack_aug(const float* __restrict__ src, const float* __restrict__ e0,
                         const float* __restrict__ e1, const float* __restrict__ e2,
                         __bf16* __restrict__ dst) {
  size_t i = (size_t)blockIdx.x * 256 + threadIdx.x;
  const size_t total = (size_t)Bb * SPp * Hdim;
  if (i >= total) return;
  int hc = (int)(i % Hdim);
  size_t rs = i / Hdim;
  int s = (int)(rs % SPp);
  int b = (int)(rs / SPp);
  float v;
  if (s < Ss)            v = src[((size_t)b * Ss + s) * Hdim + hc];
  else if (s == Ss)      v = e0[hc];
  else if (s == Ss + 1)  v = e1[hc];
  else if (s == Ss + 2)  v = e2[hc];
  else                   v = 0.0f;
  dst[i] = (__bf16)v;
}

__global__ void conv_bf16(const float* __restrict__ x, __bf16* __restrict__ y, int n) {
  int i = blockIdx.x * 256 + threadIdx.x;
  if (i < n) y[i] = (__bf16)x[i];
}

// ---------------- projection GEMM: Y = X @ W^T + bias (bf16 in/out, f32 accum) ----------------
// Ping-pong software pipeline (2x unrolled K loop): loads write straight into the register
// set they are consumed from -> no rotation copies, loads overlap WMMAs.
__global__ __launch_bounds__(256) void proj_kernel(const __bf16* __restrict__ X,
                                                   const __bf16* __restrict__ W,
                                                   const float* __restrict__ bias,
                                                   __bf16* __restrict__ Y) {
  const int tid = threadIdx.x, wave = tid >> 5, lane = tid & 31;
  const int ln = lane & 15, hl = lane >> 4;
  const int mt = blockIdx.x % (SPp / 16);
  const int b  = blockIdx.x / (SPp / 16);
  const int n0 = blockIdx.y * 512 + wave * 64;
  const __bf16* xp = X + ((size_t)b * SPp + mt * 16 + ln) * Hdim;
  const __bf16* w0 = W + (size_t)(n0 +  0 + ln) * Hdim;
  const __bf16* w1 = W + (size_t)(n0 + 16 + ln) * Hdim;
  const __bf16* w2 = W + (size_t)(n0 + 32 + ln) * Hdim;
  const __bf16* w3 = W + (size_t)(n0 + 48 + ln) * Hdim;
  v8f c0 = {}, c1 = {}, c2 = {}, c3 = {};
  // set 0 (even steps)
  v16bf a0  = load_frag(xp, hl);
  v16bf p0  = load_frag(w0, hl);
  v16bf p1  = load_frag(w1, hl);
  v16bf p2  = load_frag(w2, hl);
  v16bf p3  = load_frag(w3, hl);
  for (int k0 = 0; k0 < Hdim; k0 += 64) {
    // set 1 (odd step) loads overlap set-0 WMMAs
    v16bf a1 = load_frag(xp + k0 + 32, hl);
    v16bf q0 = load_frag(w0 + k0 + 32, hl);
    v16bf q1 = load_frag(w1 + k0 + 32, hl);
    v16bf q2 = load_frag(w2 + k0 + 32, hl);
    v16bf q3 = load_frag(w3 + k0 + 32, hl);
    c0 = wmma_bf(a0, p0, c0);
    c1 = wmma_bf(a0, p1, c1);
    c2 = wmma_bf(a0, p2, c2);
    c3 = wmma_bf(a0, p3, c3);
    if (k0 + 64 < Hdim) {  // reload set 0 for next even step, overlapping set-1 WMMAs
      a0 = load_frag(xp + k0 + 64, hl);
      p0 = load_frag(w0 + k0 + 64, hl);
      p1 = load_frag(w1 + k0 + 64, hl);
      p2 = load_frag(w2 + k0 + 64, hl);
      p3 = load_frag(w3 + k0 + 64, hl);
    }
    c0 = wmma_bf(a1, q0, c0);
    c1 = wmma_bf(a1, q1, c1);
    c2 = wmma_bf(a1, q2, c2);
    c3 = wmma_bf(a1, q3, c3);
  }
  const size_t rowbase = (size_t)b * SPp + mt * 16;
#pragma unroll
  for (int t = 0; t < 4; ++t) {
    const v8f c = (t == 0) ? c0 : (t == 1) ? c1 : (t == 2) ? c2 : c3;
    const int n = n0 + t * 16 + ln;
    const float bt = bias[n];
#pragma unroll
    for (int r = 0; r < 8; ++r)
      Y[(rowbase + r + 8 * hl) * Hdim + n] = (__bf16)(c[r] + bt);
  }
}

// ---------------- flash attention: softmax(QK^T/8) @ V per (b,h), bf16 WMMA ----------------
__global__ __launch_bounds__(192) void attn_kernel(const __bf16* __restrict__ Q,
                                                   const __bf16* __restrict__ Km,
                                                   const __bf16* __restrict__ Vm,
                                                   __bf16* __restrict__ att) {
  __shared__ __bf16 Kl[32][64];        // K tile row-major: [key][d] (shared by all 6 waves)
  __shared__ __bf16 Vt[64][32];        // V tile transposed: [d][key]
  __shared__ __bf16 Pw[6][16][32];     // per-wave probability tile (C-layout -> A-layout bridge)
  const int tid = threadIdx.x, wave = tid >> 5, lane = tid & 31;
  const int ln = lane & 15, hl = lane >> 4;
  const int g = blockIdx.x, h = blockIdx.y, b = blockIdx.z;
  const int qt = g * 6 + wave;                    // 0..65
  const size_t qrow0 = (size_t)b * SPp + qt * 16;
  const __bf16* qbase = Q + (qrow0 + ln) * Hdim + h * HDh;
  const v16bf aQ0 = load_frag(qbase, hl);         // d 0..31
  const v16bf aQ1 = load_frag(qbase + 32, hl);    // d 32..63
  v8f acc0 = {}, acc1 = {}, acc2 = {}, acc3 = {};
  float mrow[8], lrow[8];
#pragma unroll
  for (int r = 0; r < 8; ++r) { mrow[r] = -1e30f; lrow[r] = 0.0f; }
  const float scale = 0.125f;  // 1/sqrt(64)

  for (int kk = 0; kk < SPp / 32; ++kk) {
    const int key0 = kk * 32;
    __syncthreads();
    // ---- stage K tile into LDS (straight copy: async DMA path if available) ----
    for (int t = tid; t < 256; t += 192) {
      const int k = t >> 3, d0 = (t & 7) * 8;
      const __bf16* gp = Km + ((size_t)b * SPp + key0 + k) * Hdim + h * HDh + d0;
#ifdef USE_ASYNC_LDS
      __builtin_amdgcn_global_load_async_to_lds_b128(
          (__attribute__((address_space(1))) v4i*)gp,
          (__attribute__((address_space(3))) v4i*)&Kl[k][d0], 0, 0);
#else
      *reinterpret_cast<v8bf*>(&Kl[k][d0]) = *reinterpret_cast<const v8bf*>(gp);
#endif
    }
    // ---- stage V^T tile into LDS (transpose: vector global loads, strided ds stores) ----
    for (int t = tid; t < 256; t += 192) {
      const int k = t >> 3, d0 = (t & 7) * 8;
      v8bf v = *reinterpret_cast<const v8bf*>(
          Vm + ((size_t)b * SPp + key0 + k) * Hdim + h * HDh + d0);
#pragma unroll
      for (int i = 0; i < 8; ++i) Vt[d0 + i][k] = v[i];
    }
    if (key0 + 32 < SPp)  // prefetch next K tile toward L2
      __builtin_prefetch(Km + ((size_t)b * SPp + key0 + 32) * Hdim + h * HDh, 0, 0);
#ifdef USE_ASYNC_LDS
    __builtin_amdgcn_s_wait_asynccnt(0);
#endif
    __syncthreads();

    // ---- scores for two 16-key tiles (K fragments from LDS, shared by all waves) ----
    const __bf16* kbaseA = &Kl[ln][0];
    const __bf16* kbaseB = &Kl[16 + ln][0];
    v8f sA = {}, sB = {};
    sA = wmma_bf(aQ0, load_frag(kbaseA, hl), sA);
    sA = wmma_bf(aQ1, load_frag(kbaseA + 32, hl), sA);
    sB = wmma_bf(aQ0, load_frag(kbaseB, hl), sB);
    sB = wmma_bf(aQ1, load_frag(kbaseB + 32, hl), sB);

    const bool maskA = (key0 + ln) >= SAa;
    const bool maskB = (key0 + 16 + ln) >= SAa;
    float pAv[8], pBv[8];
#pragma unroll
    for (int r = 0; r < 8; ++r) {
      float a  = maskA ? -1e30f : sA[r] * scale;
      float b2 = maskB ? -1e30f : sB[r] * scale;
      float rm = fmaxf(rmax16(a), rmax16(b2));
      float mnew = fmaxf(mrow[r], rm);
      float alpha = __expf(mrow[r] - mnew);
      float pa = __expf(a - mnew);
      float pb = __expf(b2 - mnew);
      lrow[r] = lrow[r] * alpha + rsum16(pa) + rsum16(pb);
      mrow[r] = mnew;
      acc0[r] *= alpha; acc1[r] *= alpha; acc2[r] *= alpha; acc3[r] *= alpha;
      pAv[r] = pa; pBv[r] = pb;
    }
    // C-layout -> A-layout via per-wave LDS tile (LDS is in-order within a wave)
#pragma unroll
    for (int r = 0; r < 8; ++r) {
      Pw[wave][r + 8 * hl][ln]      = (__bf16)pAv[r];
      Pw[wave][r + 8 * hl][16 + ln] = (__bf16)pBv[r];
    }
    const v16bf aP = load_frag(&Pw[wave][ln][0], hl);
    acc0 = wmma_bf(aP, load_frag(&Vt[ 0 + ln][0], hl), acc0);
    acc1 = wmma_bf(aP, load_frag(&Vt[16 + ln][0], hl), acc1);
    acc2 = wmma_bf(aP, load_frag(&Vt[32 + ln][0], hl), acc2);
    acc3 = wmma_bf(aP, load_frag(&Vt[48 + ln][0], hl), acc3);
  }

#pragma unroll
  for (int r = 0; r < 8; ++r) {
    const float inv = 1.0f / lrow[r];
    const size_t o = ((size_t)b * SPp + qt * 16 + r + 8 * hl) * Hdim + h * HDh + ln;
    att[o +  0] = (__bf16)(acc0[r] * inv);
    att[o + 16] = (__bf16)(acc1[r] * inv);
    att[o + 32] = (__bf16)(acc2[r] * inv);
    att[o + 48] = (__bf16)(acc3[r] * inv);
  }
}

// ---------------- output projection + residual + LayerNorm (writes f32 out) ----------------
__global__ __launch_bounds__(256) void outln_kernel(const __bf16* __restrict__ A,
                                                    const __bf16* __restrict__ Wo,
                                                    const float* __restrict__ bo,
                                                    const float* __restrict__ Xres,
                                                    const float* __restrict__ gam,
                                                    const float* __restrict__ bet,
                                                    float* __restrict__ out) {
  __shared__ float redS[8][16];
  __shared__ float redQ[8][16];
  const int tid = threadIdx.x, wave = tid >> 5, lane = tid & 31;
  const int ln = lane & 15, hl = lane >> 4;
  const int mt = blockIdx.x % (Ss / 16);
  const int b  = blockIdx.x / (Ss / 16);
  const int n0 = wave * 128;
  const __bf16* ap = A + ((size_t)b * SPp + mt * 16 + ln) * Hdim;
  v8f c[8] = {};
  v16bf a0 = load_frag(ap, hl);
  for (int k0 = 0; k0 < Hdim; k0 += 64) {   // ping-pong on the A fragment
    v16bf a1 = load_frag(ap + k0 + 32, hl);
#pragma unroll
    for (int t = 0; t < 8; ++t)
      c[t] = wmma_bf(a0, load_frag(Wo + (size_t)(n0 + t * 16 + ln) * Hdim + k0, hl), c[t]);
    if (k0 + 64 < Hdim) a0 = load_frag(ap + k0 + 64, hl);
#pragma unroll
    for (int t = 0; t < 8; ++t)
      c[t] = wmma_bf(a1, load_frag(Wo + (size_t)(n0 + t * 16 + ln) * Hdim + k0 + 32, hl), c[t]);
  }
  // bias + residual (in place)
#pragma unroll
  for (int t = 0; t < 8; ++t) {
    const int n = n0 + t * 16 + ln;
    const float bt = bo[n];
#pragma unroll
    for (int r = 0; r < 8; ++r) {
      const int srow = mt * 16 + r + 8 * hl;
      c[t][r] = c[t][r] + bt + Xres[((size_t)b * Ss + srow) * Hdim + n];
    }
  }
  // per-row partials over this wave's 128 columns
#pragma unroll
  for (int r = 0; r < 8; ++r) {
    float s = 0.f, q = 0.f;
#pragma unroll
    for (int t = 0; t < 8; ++t) { float v = c[t][r]; s += v; q += v * v; }
    s = rsum16(s); q = rsum16(q);
    if (ln == 0) { redS[wave][r + 8 * hl] = s; redQ[wave][r + 8 * hl] = q; }
  }
  __syncthreads();
  float mu[8], rs[8];
#pragma unroll
  for (int r = 0; r < 8; ++r) {
    const int row = r + 8 * hl;
    float s = 0.f, q = 0.f;
#pragma unroll
    for (int w = 0; w < 8; ++w) { s += redS[w][row]; q += redQ[w][row]; }
    const float m = s * (1.0f / Hdim);
    mu[r] = m;
    rs[r] = rsqrtf(q * (1.0f / Hdim) - m * m + 1e-5f);
  }
#pragma unroll
  for (int t = 0; t < 8; ++t) {
    const int n = n0 + t * 16 + ln;
    const float gg = gam[n], bb = bet[n];
#pragma unroll
    for (int r = 0; r < 8; ++r) {
      const int srow = mt * 16 + r + 8 * hl;
      out[((size_t)b * Ss + srow) * Hdim + n] = (c[t][r] - mu[r]) * rs[r] * gg + bb;
    }
  }
}

extern "C" void kernel_launch(void* const* d_in, const int* in_sizes, int n_in,
                              void* d_out, int out_size, void* d_ws, size_t ws_size,
                              hipStream_t stream) {
  (void)in_sizes; (void)n_in; (void)out_size; (void)ws_size;
  const float* cnn = (const float*)d_in[0];
  const float* llm = (const float*)d_in[1];
  const float* Wq  = (const float*)d_in[2];  const float* bq  = (const float*)d_in[3];
  const float* Wk  = (const float*)d_in[4];  const float* bk  = (const float*)d_in[5];
  const float* Wv  = (const float*)d_in[6];  const float* bv  = (const float*)d_in[7];
  const float* Wo  = (const float*)d_in[8];  const float* bo  = (const float*)d_in[9];
  const float* gam = (const float*)d_in[10]; const float* bet = (const float*)d_in[11];
  const float* e0  = (const float*)d_in[12];
  const float* e1  = (const float*)d_in[13];
  const float* e2  = (const float*)d_in[14];
  float* out = (float*)d_out;

  // workspace carve-up (bf16 buffers, 256B aligned)
  char* cur = (char*)d_ws;
  auto carve = [&](size_t bytes) -> __bf16* {
    __bf16* p = (__bf16*)cur;
    cur += (bytes + 255) & ~(size_t)255;
    return p;
  };
  const size_t augE = (size_t)Bb * SPp * Hdim;          // padded matrix elems
  __bf16* augA = carve(augE * 2);
  __bf16* augB = carve(augE * 2);
  __bf16* WqB  = carve((size_t)Hdim * Hdim * 2);
  __bf16* WkB  = carve((size_t)Hdim * Hdim * 2);
  __bf16* WvB  = carve((size_t)Hdim * Hdim * 2);
  __bf16* WoB  = carve((size_t)Hdim * Hdim * 2);
  __bf16* Qa = carve(augE * 2); __bf16* Ka = carve(augE * 2); __bf16* Va = carve(augE * 2);
  __bf16* Qb = carve(augE * 2); __bf16* Kb = carve(augE * 2); __bf16* Vb = carve(augE * 2);
  __bf16* at1 = carve(augE * 2);
  __bf16* at2 = carve(augE * 2);

  const int packBlocks = (int)((augE + 255) / 256);
  pack_aug<<<packBlocks, 256, 0, stream>>>(cnn, e0, e1, e2, augA);
  pack_aug<<<packBlocks, 256, 0, stream>>>(llm, e0, e1, e2, augB);
  const int wn = Hdim * Hdim;
  conv_bf16<<<(wn + 255) / 256, 256, 0, stream>>>(Wq, WqB, wn);
  conv_bf16<<<(wn + 255) / 256, 256, 0, stream>>>(Wk, WkB, wn);
  conv_bf16<<<(wn + 255) / 256, 256, 0, stream>>>(Wv, WvB, wn);
  conv_bf16<<<(wn + 255) / 256, 256, 0, stream>>>(Wo, WoB, wn);

  dim3 pg(Bb * (SPp / 16), 2);
  proj_kernel<<<pg, 256, 0, stream>>>(augA, WqB, bq, Qa);
  proj_kernel<<<pg, 256, 0, stream>>>(augA, WkB, bk, Ka);
  proj_kernel<<<pg, 256, 0, stream>>>(augA, WvB, bv, Va);
  proj_kernel<<<pg, 256, 0, stream>>>(augB, WqB, bq, Qb);
  proj_kernel<<<pg, 256, 0, stream>>>(augB, WkB, bk, Kb);
  proj_kernel<<<pg, 256, 0, stream>>>(augB, WvB, bv, Vb);

  dim3 ag(SPp / 16 / 6, NHEAD, Bb);  // 11 x 16 x 4
  attn_kernel<<<ag, 192, 0, stream>>>(Qa, Kb, Vb, at1);  // cnn attends over llm
  attn_kernel<<<ag, 192, 0, stream>>>(Qb, Ka, Va, at2);  // llm attends over cnn

  const size_t outHalf = (size_t)Bb * Ss * Hdim;
  outln_kernel<<<Bb * (Ss / 16), 256, 0, stream>>>(at1, WoB, bo, cnn, gam, bet, out);
  outln_kernel<<<Bb * (Ss / 16), 256, 0, stream>>>(at2, WoB, bo, llm, gam, bet, out + outHalf);
}
